// VGG16Morlet_61675730371073
// MI455X (gfx1250) — matrix operand order repaired and checked by reading, not compiled, not run on hardware
//
#include <hip/hip_runtime.h>
#include <hip/hip_bf16.h>
#include <cstdint>

typedef __attribute__((ext_vector_type(16))) _Float16 v16h;
typedef __attribute__((ext_vector_type(8)))  float    v8f;
typedef __attribute__((ext_vector_type(8)))  _Float16 h8;
typedef __attribute__((ext_vector_type(4)))  unsigned int u32x4;
typedef __attribute__((ext_vector_type(8)))  int i32x8;
typedef __attribute__((ext_vector_type(4)))  int i32x4;

#if defined(__HIP_DEVICE_COMPILE__) && __has_builtin(__builtin_amdgcn_tensor_load_to_lds) && __has_builtin(__builtin_amdgcn_s_wait_tensorcnt)
#define HAVE_TDM 1
#else
#define HAVE_TDM 0
#endif

// ---------------------------------------------------------------------------
// TDM: stage a [rows][k<=32] f16 tile (row stride = rowStrideElems) to LDS.
// D# layout per CDNA5 ISA (2D, data_size=2B).
// ---------------------------------------------------------------------------
#if HAVE_TDM
__device__ __forceinline__ void tdm_load_b_tile(const _Float16* gptr,
                                                unsigned lds_off,
                                                unsigned rowStrideElems,
                                                unsigned kValid,
                                                unsigned rows) {
  unsigned long long ga = (unsigned long long)(uintptr_t)gptr;
  unsigned td0 = kValid;   // tensor_dim0: zero-fill reads past valid k
  unsigned td1 = rows;     // tensor_dim1
  unsigned tile0 = 32u, tile1 = rows;
  unsigned long long st = (unsigned long long)rowStrideElems;
  u32x4 g0;
  g0[0] = 1u;                                            // count=1 (valid)
  g0[1] = lds_off;                                       // lds_addr (bytes)
  g0[2] = (unsigned)(ga & 0xFFFFFFFFu);                  // global_addr[31:0]
  g0[3] = (unsigned)((ga >> 32) & 0x01FFFFFFu) | (2u << 30); // addr[56:32] | type=2
  i32x8 g1;
  g1[0] = (int)(1u << 16);                               // data_size = 2B
  g1[1] = (int)((td0 & 0xFFFFu) << 16);                  // tensor_dim0[15:0]
  g1[2] = (int)((td0 >> 16) | ((td1 & 0xFFFFu) << 16));  // td0[31:16] | td1[15:0]
  g1[3] = (int)((td1 >> 16) | (tile0 << 16));            // td1[31:16] | tile_dim0
  g1[4] = (int)(tile1 & 0xFFFFu);                        // tile_dim1 (tile_dim2=0)
  g1[5] = (int)(st & 0xFFFFFFFFu);                       // tensor_dim0_stride lo
  g1[6] = (int)((st >> 32) & 0xFFFFu);                   // stride hi (dim1_stride=0)
  g1[7] = 0;
  i32x4 z4 = {0, 0, 0, 0};
#if __clang_major__ >= 23
  i32x8 z8 = {0, 0, 0, 0, 0, 0, 0, 0};
  __builtin_amdgcn_tensor_load_to_lds(g0, g1, z4, z4, z8, 0);
#else
  __builtin_amdgcn_tensor_load_to_lds(g0, g1, z4, z4, 0);
#endif
}
__device__ __forceinline__ void tdm_wait() { __builtin_amdgcn_s_wait_tensorcnt(0); }
#endif

// ---------------------------------------------------------------------------
// Stage 1: Morlet integrated wavelet table (compile-time numpy const in ref)
// ---------------------------------------------------------------------------
__global__ void init_int_psi(float* __restrict__ int_psi) {
  if (threadIdx.x == 0 && blockIdx.x == 0) {
    const float step = 16.0f / 1023.0f;
    float acc = 0.f;
    for (int i = 0; i < 1024; ++i) {
      float xp = -8.f + step * (float)i;
      float psi = expf(-xp * xp * 0.5f) * cosf(5.f * xp);
      acc += psi;
      int_psi[i] = acc * step;
    }
  }
}

__global__ void cwt_kernel(const float* __restrict__ x,
                           const float* __restrict__ int_psi,
                           float* __restrict__ coeffs) {
  int idx = blockIdx.x * blockDim.x + threadIdx.x;
  if (idx >= 32 * 23 * 400) return;
  int b = idx / 9200;
  int r = idx % 9200;
  int sIdx = r / 400, t = r % 400;
  int s = sIdx + 1;
  int K = 16 * s + 1;
  int center = t + 8 * s;
  int den = 16 * s;
  const float* xb = x + b * 400;
  int i0 = center - 399; if (i0 < 0) i0 = 0;
  int i1 = center < K ? center : K;
  float acc = 0.f;
  for (int i = i0; i <= i1; ++i) {
    float kji  = (i < K)  ? int_psi[((K - 1 - i) * 1023) / den] : 0.f;
    float kjm1 = (i >= 1) ? int_psi[((K - i) * 1023) / den]     : 0.f;
    acc += xb[center - i] * (kji - kjm1);
  }
  coeffs[idx] = -sqrtf((float)s) * acc;
}

__global__ void maxabs_kernel(const float* __restrict__ coeffs,
                              float* __restrict__ maxv) {
  int b = blockIdx.x;
  const float* p = coeffs + (size_t)b * 9200;
  float mx = 0.f;
  for (int i = threadIdx.x; i < 9200; i += 256) mx = fmaxf(mx, fabsf(p[i]));
  __shared__ float sm[256];
  sm[threadIdx.x] = mx;
  __syncthreads();
  for (int st = 128; st > 0; st >>= 1) {
    if (threadIdx.x < st) sm[threadIdx.x] = fmaxf(sm[threadIdx.x], sm[threadIdx.x + st]);
    __syncthreads();
  }
  if (threadIdx.x == 0) maxv[b] = sm[0];
}

// ---------------------------------------------------------------------------
// Stage 2: colormap + half-pixel bilinear resize -> NHWC f16 [32,224,224,3]
// ---------------------------------------------------------------------------
__global__ void resize_cmap(const float* __restrict__ coeffs,
                            const float* __restrict__ maxv,
                            const float* __restrict__ lut,
                            _Float16* __restrict__ img) {
  int idx = blockIdx.x * blockDim.x + threadIdx.x;
  if (idx >= 32 * 224 * 224) return;
  int b = idx / (224 * 224);
  int r = idx % (224 * 224);
  int oy = r / 224, ox = r % 224;
  float m = maxv[b];
  float fy = (oy + 0.5f) * (23.f / 224.f) - 0.5f;
  float fx = (ox + 0.5f) * (400.f / 224.f) - 0.5f;
  int y0 = (int)floorf(fy); float wy = fy - (float)y0;
  int x0 = (int)floorf(fx); float wx = fx - (float)x0;
  int y0c = min(max(y0, 0), 22),  y1c = min(max(y0 + 1, 0), 22);
  int x0c = min(max(x0, 0), 399), x1c = min(max(x0 + 1, 0), 399);
  const float* cb = coeffs + (size_t)b * 9200;
  auto q = [&](int y, int x) -> int {
    float c = cb[y * 400 + x];
    float nv = (m > 0.f) ? (0.5f + 0.5f * c / fmaxf(m, 1e-30f)) : c;
    int qi = (int)(nv * 256.f);
    return min(max(qi, 0), 255);
  };
  int q00 = q(y0c, x0c), q01 = q(y0c, x1c), q10 = q(y1c, x0c), q11 = q(y1c, x1c);
  _Float16* o = img + (size_t)idx * 3;
#pragma unroll
  for (int ch = 0; ch < 3; ++ch) {
    float v00 = lut[q00 * 3 + ch], v01 = lut[q01 * 3 + ch];
    float v10 = lut[q10 * 3 + ch], v11 = lut[q11 * 3 + ch];
    float top = v00 + (v01 - v00) * wx;
    float bot = v10 + (v11 - v10) * wx;
    o[ch] = (_Float16)(top + (bot - top) * wy);
  }
}

// ---------------------------------------------------------------------------
// Weight prep: OIHW f32 -> [Cout][(kh*3+kw)*Cin + cin] f16
// ---------------------------------------------------------------------------
__global__ void conv_w_cvt(const float* __restrict__ w, _Float16* __restrict__ o,
                           int CIN, int COUT) {
  int idx = blockIdx.x * blockDim.x + threadIdx.x;
  int total = COUT * CIN * 9;
  if (idx >= total) return;
  int oc = idx / (CIN * 9);
  int r = idx % (CIN * 9);
  int ci = r / 9;
  int kk = r % 9;
  o[(size_t)oc * 9 * CIN + kk * CIN + ci] = (_Float16)w[idx];
}

// ---------------------------------------------------------------------------
// Implicit-GEMM 3x3 SAME conv, NHWC f16, f32 accum, TDM-staged B tiles.
// Block = 4 waves = 64(M) x 64(N); wave = 16(M) x 64(N) -> 4 WMMAs / K-step,
// one shared A fragment. B tile (64x32 f16) double-buffered in LDS via TDM.
// All row decompositions are 32-bit with incremental wrap (no 64-bit div).
// ---------------------------------------------------------------------------
template <int CIN>
__global__ void conv3x3_wmma(const _Float16* __restrict__ in,
                             const _Float16* __restrict__ wgt,   // [COUT][9*CIN]
                             const float* __restrict__ bias,
                             _Float16* __restrict__ out,
                             int H, int W, int COUT) {
  const int K = 9 * CIN;
  const int KSTEPS = (K + 31) / 32;
  __shared__ __align__(16) _Float16 lbuf[2][64 * 32];

  const int lane = threadIdx.x & 31;
  const int wv   = threadIdx.x >> 5;
  const int half = lane >> 4;
  const int n0 = blockIdx.y * 64;
  const unsigned HW = (unsigned)(H * W);

  // one 32-bit div/mod pair for the wave's 16-row tile base
  const unsigned m0 = ((unsigned)blockIdx.x * 4u + (unsigned)wv) * 16u;
  const unsigned nb0 = m0 / HW;
  const unsigned rem0 = m0 - nb0 * HW;
  const unsigned oh0 = rem0 / (unsigned)W;
  const unsigned ow0 = rem0 - oh0 * (unsigned)W;

  // A-row for this lane: advance by (lane&15) with incremental wrap
  int nb = (int)nb0, oh = (int)oh0, ow = (int)ow0 + (lane & 15);
  while (ow >= W) { ow -= W; ++oh; }
  if (oh >= H) { oh -= H; ++nb; }
  const _Float16* inbase = in + (size_t)nb * HW * CIN;
  const _Float16* wrow = wgt + (size_t)n0 * K;

#if HAVE_TDM
  if (wv == 0) {
    tdm_load_b_tile(wrow, (unsigned)(uintptr_t)&lbuf[0][0], (unsigned)K,
                    (unsigned)(K < 32 ? K : 32), 64u);
    tdm_wait();
  }
  __syncthreads();
#endif

  v8f c0 = {}, c1 = {}, c2 = {}, c3 = {};
  for (int ks = 0; ks < KSTEPS; ++ks) {
    const int kk0 = ks * 32;
    const int cur = ks & 1;
#if HAVE_TDM
    if (wv == 0 && ks + 1 < KSTEPS) {
      int nk = kk0 + 32;
      tdm_load_b_tile(wrow + nk, (unsigned)(uintptr_t)&lbuf[1 - cur][0], (unsigned)K,
                      (unsigned)((K - nk) < 32 ? (K - nk) : 32), 64u);
    }
#else
    {  // cooperative fallback staging: 128 threads x 16 halves = 64x32 tile
      __syncthreads();
      int rowi = threadIdx.x >> 1;
      int ko   = (threadIdx.x & 1) * 16;
#pragma unroll
      for (int e = 0; e < 16; ++e) {
        int k = kk0 + ko + e;
        lbuf[cur][rowi * 32 + ko + e] =
            (k < K) ? wrow[(size_t)rowi * K + k] : (_Float16)0.f;
      }
      __syncthreads();
    }
#endif
    // ---- A fragment ----
    v16h a;
    if (CIN == 3) {  // first layer: element-wise zero-padded im2col gather
#pragma unroll
      for (int e = 0; e < 16; ++e) {
        int vg = e >> 1, sub = e & 1;
        int k = kk0 + (vg < 4 ? 2 * vg : 16 + 2 * (vg - 4)) + 8 * half + sub;
        _Float16 val = (_Float16)0.f;
        if (k < K) {
          int khw = k / CIN;
          int cin = k - khw * CIN;
          int kh = khw / 3, kw = khw - kh * 3;
          int ih = oh + kh - 1, iw = ow + kw - 1;
          if ((unsigned)ih < (unsigned)H && (unsigned)iw < (unsigned)W)
            val = inbase[((size_t)ih * W + iw) * CIN + cin];
        }
        a[e] = val;
      }
    } else {  // CIN % 32 == 0: step lies in one (kh,kw) plane -> 2x b128
      int khw = kk0 / CIN;
      int cinb = kk0 - khw * CIN;
      int kh = khw / 3, kw = khw - kh * 3;
      int ih = oh + kh - 1, iw = ow + kw - 1;
      bool inb = (unsigned)ih < (unsigned)H && (unsigned)iw < (unsigned)W;
      h8 a0 = {}, a1 = {};
      if (inb) {
        const _Float16* p = inbase + ((size_t)ih * W + iw) * CIN + cinb + 8 * half;
        a0 = *(const h8*)p;
        a1 = *(const h8*)(p + 16);
      }
#pragma unroll
      for (int e = 0; e < 8; ++e) { a[e] = a0[e]; a[e + 8] = a1[e]; }
    }
    // ---- B fragments from LDS (ds_load_b128), 4 N-blocks ----
    v16h b0, b1, b2, b3;
    {
      const h8* p0 = (const h8*)&lbuf[cur][((lane & 15) +  0) * 32 + 8 * half];
      const h8* p1 = (const h8*)&lbuf[cur][((lane & 15) + 16) * 32 + 8 * half];
      const h8* p2 = (const h8*)&lbuf[cur][((lane & 15) + 32) * 32 + 8 * half];
      const h8* p3 = (const h8*)&lbuf[cur][((lane & 15) + 48) * 32 + 8 * half];
      h8 x0 = p0[0], x1 = p0[2], y0 = p1[0], y1 = p1[2];
      h8 z0 = p2[0], z1 = p2[2], w0 = p3[0], w1 = p3[2];
#pragma unroll
      for (int e = 0; e < 8; ++e) {
        b0[e] = x0[e]; b0[e + 8] = x1[e];
        b1[e] = y0[e]; b1[e + 8] = y1[e];
        b2[e] = z0[e]; b2[e + 8] = z1[e];
        b3[e] = w0[e]; b3[e + 8] = w1[e];
      }
    }
    c0 = __builtin_amdgcn_wmma_f32_16x16x32_f16(false, a, false, b0, (short)0, c0, false, false);
    c1 = __builtin_amdgcn_wmma_f32_16x16x32_f16(false, a, false, b1, (short)0, c1, false, false);
    c2 = __builtin_amdgcn_wmma_f32_16x16x32_f16(false, a, false, b2, (short)0, c2, false, false);
    c3 = __builtin_amdgcn_wmma_f32_16x16x32_f16(false, a, false, b3, (short)0, c3, false, false);
#if HAVE_TDM
    if (ks + 1 < KSTEPS) {
      if (wv == 0) tdm_wait();
      __syncthreads();
    }
#endif
  }
  // ---- epilogue: bias + ReLU, NHWC f16 (incremental row decode) ----
  const int n = n0 + (lane & 15);
  const float b0v = bias[n], b1v = bias[n + 16], b2v = bias[n + 32], b3v = bias[n + 48];
#pragma unroll
  for (int r = 0; r < 8; ++r) {
    int d = r + 8 * half;
    int nb2 = (int)nb0, oh2 = (int)oh0, ow2 = (int)ow0 + d;
    while (ow2 >= W) { ow2 -= W; ++oh2; }
    if (oh2 >= H) { oh2 -= H; ++nb2; }
    size_t obase = (((size_t)nb2 * H + oh2) * W + ow2) * COUT;
    out[obase + n]      = (_Float16)fmaxf(c0[r] + b0v, 0.f);
    out[obase + n + 16] = (_Float16)fmaxf(c1[r] + b1v, 0.f);
    out[obase + n + 32] = (_Float16)fmaxf(c2[r] + b2v, 0.f);
    out[obase + n + 48] = (_Float16)fmaxf(c3[r] + b3v, 0.f);
  }
}

__global__ void maxpool2x2(const _Float16* __restrict__ in,
                           _Float16* __restrict__ out, int H, int W, int C) {
  int OH = H / 2, OW = W / 2;
  size_t total = (size_t)32 * OH * OW * C;
  size_t idx = (size_t)blockIdx.x * blockDim.x + threadIdx.x;
  if (idx >= total) return;
  int c = (int)(idx % C);
  size_t r = idx / C;
  int ow = (int)(r % OW); r /= OW;
  int oh = (int)(r % OH);
  int n  = (int)(r / OH);
  const _Float16* p = in + (((size_t)n * H + oh * 2) * W + ow * 2) * C + c;
  float v0 = (float)p[0], v1 = (float)p[C];
  float v2 = (float)p[(size_t)W * C], v3 = (float)p[(size_t)W * C + C];
  out[idx] = (_Float16)fmaxf(fmaxf(v0, v1), fmaxf(v2, v3));
}

// ---------------------------------------------------------------------------
// FC GEMM: act f16 [32][K] x w f32 [N][K]^T + bias; wave = 16x16 tile.
// PERM=1: CHW->HWC flatten permutation on the A index (fc1 only).
// ---------------------------------------------------------------------------
template <int PERM>
__global__ void fc_wmma(const _Float16* __restrict__ act,
                        const float* __restrict__ w,
                        const float* __restrict__ bias,
                        _Float16* __restrict__ outh, float* __restrict__ outf,
                        int K, int N, int relu) {
  int tile = blockIdx.x * 4 + (threadIdx.x >> 5);
  int Ntiles = (N + 15) / 16;
  if (tile >= 2 * Ntiles) return;
  int mtile = tile & 1;
  int n0 = (tile >> 1) * 16;
  int lane = threadIdx.x & 31;
  int half = lane >> 4;
  int m = mtile * 16 + (lane & 15);
  int ncol = n0 + (lane & 15);
  int nclamp = ncol < N ? ncol : (N - 1);
  const float* wrow = w + (size_t)nclamp * K;
  v8f c0 = {};
  for (int kk0 = 0; kk0 < K; kk0 += 32) {
    v16h a, b;
    if (PERM) {
#pragma unroll
      for (int e = 0; e < 16; ++e) {
        int vg = e >> 1, sub = e & 1;
        int k = kk0 + (vg < 4 ? 2 * vg : 16 + 2 * (vg - 4)) + 8 * half + sub;
        int ki = (k % 49) * 512 + k / 49;  // CHW -> HWC
        a[e] = act[(size_t)m * K + ki];
      }
    } else {
      const h8* pa0 = (const h8*)(act + (size_t)m * K + kk0 + 8 * half);
      const h8* pa1 = (const h8*)(act + (size_t)m * K + kk0 + 16 + 8 * half);
      h8 a0 = pa0[0], a1 = pa1[0];
#pragma unroll
      for (int e = 0; e < 8; ++e) { a[e] = a0[e]; a[e + 8] = a1[e]; }
    }
    const float* pw = wrow + kk0 + 8 * half;
#pragma unroll
    for (int e = 0; e < 16; ++e) b[e] = (_Float16)pw[e + (e & 8)];
    c0 = __builtin_amdgcn_wmma_f32_16x16x32_f16(false, a, false, b, (short)0, c0, false, false);
  }
#pragma unroll
  for (int r = 0; r < 8; ++r) {
    int mm = mtile * 16 + r + 8 * half;
    int nn = n0 + (lane & 15);
    if (nn < N) {
      float v = c0[r] + bias[nn];
      if (relu) v = fmaxf(v, 0.f);
      if (outf) outf[(size_t)mm * N + nn] = v;
      else      outh[(size_t)mm * N + nn] = (_Float16)v;
    }
  }
}

// ---------------------------------------------------------------------------
// Host orchestration
// ---------------------------------------------------------------------------
extern "C" void kernel_launch(void* const* d_in, const int* in_sizes, int n_in,
                              void* d_out, int out_size, void* d_ws, size_t ws_size,
                              hipStream_t stream) {
  (void)in_sizes; (void)n_in; (void)out_size; (void)ws_size;
  const float* x_in = (const float*)d_in[0];
  const float* lut  = (const float*)d_in[1];

  static const int cins[13]  = {3, 64, 64, 128, 128, 256, 256, 256, 512, 512, 512, 512, 512};
  static const int couts[13] = {64, 64, 128, 128, 256, 256, 256, 512, 512, 512, 512, 512, 512};
  static const int hs[13]    = {224, 224, 112, 112, 56, 56, 56, 28, 28, 28, 14, 14, 14};
  static const bool poolAfter[13] = {false, true, false, true, false, false, true,
                                     false, false, true, false, false, true};

  char* ws = (char*)d_ws;
  size_t cur = 0;
  auto alloc = [&](size_t bytes) -> size_t {
    size_t o = cur;
    cur = (cur + bytes + 255) & ~(size_t)255;
    return o;
  };
  size_t off_psi    = alloc(1024 * sizeof(float));
  size_t off_coeffs = alloc((size_t)32 * 23 * 400 * sizeof(float));
  size_t off_maxv   = alloc(32 * sizeof(float));
  size_t off_w16[13];
  for (int l = 0; l < 13; ++l)
    off_w16[l] = alloc((size_t)couts[l] * cins[l] * 9 * sizeof(_Float16));
  size_t actBytes = (size_t)32 * 224 * 224 * 64 * sizeof(_Float16);
  size_t off_actA = alloc(actBytes);
  size_t off_actB = alloc(actBytes);

  float*    int_psi = (float*)(ws + off_psi);
  float*    coeffs  = (float*)(ws + off_coeffs);
  float*    maxv    = (float*)(ws + off_maxv);
  _Float16* actA    = (_Float16*)(ws + off_actA);
  _Float16* actB    = (_Float16*)(ws + off_actB);

  // --- scalogram ---
  init_int_psi<<<1, 64, 0, stream>>>(int_psi);
  cwt_kernel<<<(32 * 23 * 400 + 255) / 256, 256, 0, stream>>>(x_in, int_psi, coeffs);
  maxabs_kernel<<<32, 256, 0, stream>>>(coeffs, maxv);
  resize_cmap<<<(32 * 224 * 224 + 255) / 256, 256, 0, stream>>>(coeffs, maxv, lut, actA);

  // --- convert conv weights to f16 [Cout][9*Cin] ---
  for (int l = 0; l < 13; ++l) {
    const float* w = (const float*)d_in[2 + 2 * l];
    int total = couts[l] * cins[l] * 9;
    conv_w_cvt<<<(total + 255) / 256, 256, 0, stream>>>(w, (_Float16*)(ws + off_w16[l]),
                                                        cins[l], couts[l]);
  }

  // --- VGG16 feature stack (ping-pong NHWC f16) ---
  _Float16* curA = actA;
  _Float16* nxtA = actB;
  for (int l = 0; l < 13; ++l) {
    int H = hs[l], W = hs[l], CIN = cins[l], COUT = couts[l];
    const float* bias = (const float*)d_in[3 + 2 * l];
    const _Float16* wgt = (const _Float16*)(ws + off_w16[l]);
    long M = 32L * H * W;
    dim3 grid((unsigned)(M / 64), (unsigned)(COUT / 64));
    switch (CIN) {
      case 3:   conv3x3_wmma<3>  <<<grid, 128, 0, stream>>>(curA, wgt, bias, nxtA, H, W, COUT); break;
      case 64:  conv3x3_wmma<64> <<<grid, 128, 0, stream>>>(curA, wgt, bias, nxtA, H, W, COUT); break;
      case 128: conv3x3_wmma<128><<<grid, 128, 0, stream>>>(curA, wgt, bias, nxtA, H, W, COUT); break;
      case 256: conv3x3_wmma<256><<<grid, 128, 0, stream>>>(curA, wgt, bias, nxtA, H, W, COUT); break;
      case 512: conv3x3_wmma<512><<<grid, 128, 0, stream>>>(curA, wgt, bias, nxtA, H, W, COUT); break;
    }
    { _Float16* t = curA; curA = nxtA; nxtA = t; }
    if (poolAfter[l]) {
      size_t total = (size_t)32 * (H / 2) * (W / 2) * COUT;
      maxpool2x2<<<(unsigned)((total + 255) / 256), 256, 0, stream>>>(curA, nxtA, H, W, COUT);
      _Float16* t = curA; curA = nxtA; nxtA = t;
    }
  }
  // curA now holds NHWC f16 [32,7,7,512]

  // --- FC head ---
  {
    const float* w1 = (const float*)d_in[28];
    const float* b1 = (const float*)d_in[29];
    int K = 25088, N = 4096;
    int tiles = 2 * ((N + 15) / 16);
    fc_wmma<1><<<(tiles + 3) / 4, 128, 0, stream>>>(curA, w1, b1, nxtA, nullptr, K, N, 1);
    _Float16* t = curA; curA = nxtA; nxtA = t;
  }
  {
    const float* w2 = (const float*)d_in[30];
    const float* b2 = (const float*)d_in[31];
    int K = 4096, N = 4096;
    int tiles = 2 * ((N + 15) / 16);
    fc_wmma<0><<<(tiles + 3) / 4, 128, 0, stream>>>(curA, w2, b2, nxtA, nullptr, K, N, 1);
    _Float16* t = curA; curA = nxtA; nxtA = t;
  }
  {
    const float* w3 = (const float*)d_in[32];
    const float* b3 = (const float*)d_in[33];
    int K = 4096, N = 3;
    int tiles = 2 * ((N + 15) / 16);
    fc_wmma<0><<<(tiles + 3) / 4, 128, 0, stream>>>(curA, w3, b3, nullptr, (float*)d_out,
                                                    K, N, 0);
  }
}
